// CrossRNN_87823491269194
// MI455X (gfx1250) — matrix operand order repaired and checked by reading, not compiled, not run on hardware
//
#include <hip/hip_runtime.h>
#include <hip/hip_bf16.h>

typedef __attribute__((ext_vector_type(16))) __bf16 v16bf;
typedef __attribute__((ext_vector_type(8)))  __bf16 v8bf;
typedef __attribute__((ext_vector_type(8)))  float  v8f;

#define BB 4
#define RR 64
#define CC 64
#define SS 32
#define EE 128
#define HH 128
#define NSEQ (BB * RR * CC)
#define TILE_M 16

// ---------------------------------------------------------------------------
// WMMA helper (CDNA5: v_wmma_f32_16x16x32_bf16, f32 accumulate)
// ---------------------------------------------------------------------------
__device__ inline v8f wmma_bf16(v16bf a, v16bf b, v8f c) {
  return __builtin_amdgcn_wmma_f32_16x16x32_bf16(
      /*neg_a=*/false, a, /*neg_b=*/false, b,
      /*c_mod=*/(short)0, c, /*reuse_a=*/false, /*reuse_b=*/false);
}

// A-fragment: 16x32 bf16 chunk from LDS, row-major with rowStride (in elems).
// ISA 16-bit A layout: lanes 0-15 hold K=0..7 then 16..23, lanes 16-31 hold
// K=8..15 then 24..31, for row M = lane%16. Two contiguous 16B ds_load_b128.
__device__ inline v16bf load_a_frag(const __bf16* base, int rowStride, int k0, int lane) {
  const int lo = lane >> 4;
  const int m  = lane & 15;
  const __bf16* p = base + m * rowStride + k0 + lo * 8;
  v8bf x0 = *(const v8bf*)(p);
  v8bf x1 = *(const v8bf*)(p + 16);
  v16bf r;
#pragma unroll
  for (int i = 0; i < 8; ++i) { r[i] = x0[i]; r[i + 8] = x1[i]; }
  return r;
}

// B-fragment hoisted straight from GLOBAL memory (weights are timestep-
// invariant: load once into VGPRs, reuse for all 32 steps).
// B[k][n] = W[n][k]; the lane owning output column n reads K=8h..8h+7 then
// 16+8h..16+8h+7 (matching the A layout) and converts to bf16.
__device__ inline v16bf load_b_frag_global(const float* __restrict__ W,
                                           int n, int k0, int lane) {
  const int lo = lane >> 4;
  const float* p = W + n * HH + k0 + lo * 8;
  v16bf r;
#pragma unroll
  for (int i = 0; i < 16; ++i) r[i] = (__bf16)p[i < 8 ? i : i + 8];
  return r;
}

// Gather one timestep of embeddings (16 rows x 128 f32) into bf16 LDS tile.
__device__ inline void gather_step(const int* __restrict__ x,
                                   const float* __restrict__ embed,
                                   int seq0, int t, __bf16 (*dst)[EE], int tid) {
  const int row = tid >> 4;   // 0..15 sequence in tile
  const int seg = tid & 15;   // 8 floats per thread
  const int idx = x[(seq0 + row) * SS + t];
  const float4* src = (const float4*)(embed + (long)idx * EE + seg * 8);
  float4 a = src[0];
  float4 b = src[1];
  __bf16* d = &dst[row][seg * 8];
  d[0] = (__bf16)a.x; d[1] = (__bf16)a.y; d[2] = (__bf16)a.z; d[3] = (__bf16)a.w;
  d[4] = (__bf16)b.x; d[5] = (__bf16)b.y; d[6] = (__bf16)b.z; d[7] = (__bf16)b.w;
  // L2 prefetch one step further ahead (emits global_prefetch_b8)
  if (t + 1 < SS) {
    const int idx2 = x[(seq0 + row) * SS + t + 1];
    __builtin_prefetch(embed + (long)idx2 * EE + seg * 8, 0, 0);
  }
}

// ---------------------------------------------------------------------------
// Kernel 1: embedding gather + fused 2-layer ReLU RNN (per-timestep layer
// interleave, weights register-resident) + reduction of final hidden to two
// scalars per sequence:  s = w2 . h    t = (w1 - 2*w2) . h
// ---------------------------------------------------------------------------
__global__ __launch_bounds__(256) void crossrnn_rnn_kernel(
    const int* __restrict__ x, const float* __restrict__ embed,
    const float* __restrict__ W_ih, const float* __restrict__ W_hh,
    const float* __restrict__ b_ih, const float* __restrict__ b_hh,
    const float* __restrict__ pred_W,
    float* __restrict__ s_out, float* __restrict__ t_out) {
  __shared__ __bf16 embT[2][TILE_M][EE]; // double-buffered timestep of embeddings
  __shared__ __bf16 h1[TILE_M][HH];
  __shared__ __bf16 h2[TILE_M][HH];

  const int tid  = threadIdx.x;
  const int lane = tid & 31;
  const int wave = tid >> 5;             // 0..7 -> output columns 16*wave..+15
  const int col  = lane & 15;
  const int lo   = lane >> 4;
  const int n    = wave * 16 + col;      // global output column of this lane
  const int seq0 = blockIdx.x * TILE_M;

  // ---- hoist all weight B-fragments into registers (timestep-invariant) ----
  v16bf wi1[4], wh1[4], wi2[4], wh2[4];
#pragma unroll
  for (int kc = 0; kc < 4; ++kc) {
    const int k0 = kc * 32;
    wi1[kc] = load_b_frag_global(W_ih,            n, k0, lane);
    wh1[kc] = load_b_frag_global(W_hh,            n, k0, lane);
    wi2[kc] = load_b_frag_global(W_ih + HH * HH,  n, k0, lane);
    wh2[kc] = load_b_frag_global(W_hh + HH * HH,  n, k0, lane);
  }
  const float bias1 = b_ih[n]      + b_hh[n];
  const float bias2 = b_ih[HH + n] + b_hh[HH + n];

  // ---- zero hidden state, stage first timestep ----
  {
    __bf16* hz1 = &h1[0][0];
    __bf16* hz2 = &h2[0][0];
    for (int i = tid; i < TILE_M * HH; i += 256) {
      hz1[i] = (__bf16)0.0f;
      hz2[i] = (__bf16)0.0f;
    }
  }
  gather_step(x, embed, seq0, 0, embT[0], tid);
  __syncthreads();

  // ---- timestep loop: both layers fused per step ----
  for (int t = 0; t < SS; ++t) {
    const int buf = t & 1;
    if (t + 1 < SS) gather_step(x, embed, seq0, t + 1, embT[buf ^ 1], tid);

    // ----- layer 1: h1 = relu(emb_t @ Wi1^T + h1 @ Wh1^T + b1) -----
    v8f acc;
#pragma unroll
    for (int j = 0; j < 8; ++j) acc[j] = bias1;
#pragma unroll
    for (int kc = 0; kc < 4; ++kc) {
      const int k0 = kc * 32;
      v16bf a  = load_a_frag(&embT[buf][0][0], EE, k0, lane);
      acc = wmma_bf16(a, wi1[kc], acc);
      v16bf ah = load_a_frag(&h1[0][0], HH, k0, lane);
      acc = wmma_bf16(ah, wh1[kc], acc);
    }
    __syncthreads();  // all reads of old h1 complete
#pragma unroll
    for (int j = 0; j < 8; ++j) {
      float v = acc[j];
      h1[lo * 8 + j][n] = (__bf16)(v > 0.0f ? v : 0.0f);
    }
    __syncthreads();  // new h1 visible

    // ----- layer 2: h2 = relu(h1 @ Wi2^T + h2 @ Wh2^T + b2) -----
    v8f acc2;
#pragma unroll
    for (int j = 0; j < 8; ++j) acc2[j] = bias2;
#pragma unroll
    for (int kc = 0; kc < 4; ++kc) {
      const int k0 = kc * 32;
      v16bf a  = load_a_frag(&h1[0][0], HH, k0, lane);
      acc2 = wmma_bf16(a, wi2[kc], acc2);
      v16bf ah = load_a_frag(&h2[0][0], HH, k0, lane);
      acc2 = wmma_bf16(ah, wh2[kc], acc2);
    }
    __syncthreads();  // all reads of old h2 complete
#pragma unroll
    for (int j = 0; j < 8; ++j) {
      float v = acc2[j];
      h2[lo * 8 + j][n] = (__bf16)(v > 0.0f ? v : 0.0f);
    }
    __syncthreads();  // new h2 visible for next step
  }

  // ---- reduce final hidden to (s, t) per sequence ----
  // 16 lanes per row: each lane sums 8 hidden elements, then shuffle-reduce.
  {
    const int r = tid >> 4;  // 0..15 sequence in tile
    const int g = tid & 15;
    float s = 0.0f, tt = 0.0f;
#pragma unroll
    for (int q = 0; q < 8; ++q) {
      const int k = g * 8 + q;
      const float hv = (float)h2[r][k];
      const float w1 = pred_W[k];
      const float w2 = pred_W[HH + k];
      s  += w2 * hv;
      tt += (w1 - 2.0f * w2) * hv;
    }
#pragma unroll
    for (int off = 8; off > 0; off >>= 1) {
      s  += __shfl_xor(s, off, 16);
      tt += __shfl_xor(tt, off, 16);
    }
    if (g == 0) {
      s_out[seq0 + r] = s;
      t_out[seq0 + r] = tt;
    }
  }
}

// ---------------------------------------------------------------------------
// Kernel 2: per-batch 64x64 cross aggregation:
//   out[r][c] = t[r][c] + rowsum_s[r] + colsum_s[c] + pred_b
// ---------------------------------------------------------------------------
__global__ __launch_bounds__(256) void crossrnn_agg_kernel(
    const float* __restrict__ s, const float* __restrict__ t,
    const float* __restrict__ pred_b, float* __restrict__ out) {
  __shared__ float tile[RR][CC];
  __shared__ float rowsum[RR];
  __shared__ float colsum[CC];
  const int b   = blockIdx.x;
  const int tid = threadIdx.x;
  const float* sb = s + b * RR * CC;
  for (int i = tid; i < RR * CC; i += 256) tile[i >> 6][i & 63] = sb[i];
  __syncthreads();
  if (tid < RR) {
    float acc = 0.0f;
    for (int c = 0; c < CC; ++c) acc += tile[tid][c];
    rowsum[tid] = acc;
  } else if (tid < RR + CC) {
    const int c = tid - RR;
    float acc = 0.0f;
    for (int r = 0; r < RR; ++r) acc += tile[r][c];
    colsum[c] = acc;
  }
  __syncthreads();
  const float pb = pred_b[0];
  for (int i = tid; i < RR * CC; i += 256) {
    const int r = i >> 6, c = i & 63;
    out[b * RR * CC + i] = t[b * RR * CC + i] + rowsum[r] + colsum[c] + pb;
  }
}

// ---------------------------------------------------------------------------
extern "C" void kernel_launch(void* const* d_in, const int* in_sizes, int n_in,
                              void* d_out, int out_size, void* d_ws, size_t ws_size,
                              hipStream_t stream) {
  (void)in_sizes; (void)n_in; (void)out_size; (void)ws_size;
  const int*   x      = (const int*)d_in[0];
  const float* embed  = (const float*)d_in[1];
  const float* W_ih   = (const float*)d_in[2];
  const float* W_hh   = (const float*)d_in[3];
  const float* b_ih   = (const float*)d_in[4];
  const float* b_hh   = (const float*)d_in[5];
  const float* pred_W = (const float*)d_in[6];
  const float* pred_b = (const float*)d_in[7];

  float* s_ws = (float*)d_ws;          // NSEQ floats
  float* t_ws = s_ws + NSEQ;           // NSEQ floats

  crossrnn_rnn_kernel<<<NSEQ / TILE_M, 256, 0, stream>>>(
      x, embed, W_ih, W_hh, b_ih, b_hh, pred_W, s_ws, t_ws);
  crossrnn_agg_kernel<<<BB, 256, 0, stream>>>(s_ws, t_ws, pred_b, (float*)d_out);
}
